// GCNLayer_23356032156066
// MI455X (gfx1250) — matrix operand order repaired and checked by reading, not compiled, not run on hardware
//
#include <hip/hip_runtime.h>
#include <hip/hip_bf16.h>
#include <stddef.h>
#include <stdint.h>

#define GN 8192   // nodes
#define GD 256    // feature dim (D_IN == D_OUT)
#define BM 32     // C-tile rows per workgroup
#define BK 64     // K tile

typedef __attribute__((ext_vector_type(16))) __bf16 v16bf;
typedef __attribute__((ext_vector_type(8)))  __bf16 v8bf;
typedef __attribute__((ext_vector_type(8)))  float  v8f;
typedef __attribute__((ext_vector_type(4)))  int    v4i;

// ---- CDNA5 async global->LDS copy (ASYNCcnt path), feature-guarded ----
#if defined(__AMDGCN__) && __has_builtin(__builtin_amdgcn_global_load_async_to_lds_b128)
#define HAVE_ASYNC_LDS 1
typedef __attribute__((address_space(1))) v4i gv4i;  // global int4
typedef __attribute__((address_space(3))) v4i lv4i;  // LDS int4
__device__ __forceinline__ void async_cp16(const void* g, void* l) {
  __builtin_amdgcn_global_load_async_to_lds_b128((gv4i*)g, (lv4i*)l, 0, 0);
}
#endif

template <int N>
__device__ __forceinline__ void wait_async() {
#if defined(__AMDGCN__) && __has_builtin(__builtin_amdgcn_s_wait_asynccnt)
  __builtin_amdgcn_s_wait_asynccnt(N);
#elif defined(__AMDGCN__)
  asm volatile("s_wait_asynccnt %0" ::"n"(N) : "memory");
#endif
}

// ---------------- kernel 1: row sums of A -> d^{-1/2} ----------------
__global__ __launch_bounds__(256) void k_rowsum(const float* __restrict__ A,
                                                float* __restrict__ dinv) {
  __shared__ float red[256];
  const int row = blockIdx.x;
  const int t = threadIdx.x;
  const float* ar = A + (size_t)row * GN;
  float s = 0.f;
#pragma unroll
  for (int p = 0; p < 8; ++p) {
    float4 v = *(const float4*)(ar + p * 1024 + t * 4);
    s += v.x + v.y + v.z + v.w;
  }
  red[t] = s;
  __syncthreads();
  for (int off = 128; off > 0; off >>= 1) {
    if (t < off) red[t] += red[t + off];
    __syncthreads();
  }
  if (t == 0) dinv[row] = rsqrtf(red[0]);
}

// ---------------- kernel 2a: W -> bf16 ----------------
__global__ __launch_bounds__(256) void k_cvtW(const float* __restrict__ W,
                                              __bf16* __restrict__ Wb) {
  const int idx = blockIdx.x * 256 + threadIdx.x;
  Wb[idx] = (__bf16)W[idx];
}

// -------- kernel 2b: XsT[j,i] = bf16( X[i,j] * dinv[i] )  (scaled transpose) --------
__global__ __launch_bounds__(256) void k_scaleT(const float* __restrict__ X,
                                                const float* __restrict__ dinv,
                                                __bf16* __restrict__ XsT) {
  const size_t idx = (size_t)blockIdx.x * 256 + threadIdx.x;  // < GD*GN
  const int j = (int)(idx >> 13);       // feature (GN == 2^13)
  const int i = (int)(idx & (GN - 1));  // node
  XsT[idx] = (__bf16)(X[(size_t)i * GD + j] * dinv[i]);
}

// ---------------- kernel 3: H = diag(dinv) * (A @ Xs), bf16 out ----------------
// grid = GN/BM = 256 blocks, 256 threads = 8 waves (2 M-slices x 4 N-slices)
// Double-buffered LDS; tiles staged with async global->LDS copies when available.
__global__ __launch_bounds__(256) void k_aggr(const float* __restrict__ A,
                                              const __bf16* __restrict__ XsT,
                                              const float* __restrict__ dinv,
                                              __bf16* __restrict__ H) {
  __shared__ float  As[2][BM][BK + 4];  // 2 x 32 x 68 f32
  __shared__ __bf16 Bs[2][GD][BK + 8];  // 2 x 256 x 72 bf16
  const int tid  = threadIdx.x;
  const int lane = tid & 31;
  const int l15  = lane & 15;
  const int hi   = lane >> 4;   // lane half
  const int wid  = tid >> 5;
  const int wm   = wid & 1;     // 2 x 16 rows
  const int wn   = wid >> 1;    // 4 x 64 cols
  const int rowBase = blockIdx.x * BM;

  // per-thread staging coordinates (constant across tiles)
  const int ar0 = tid >> 4;         // A row (p=0), +16 for p=1
  const int ac  = (tid & 15) << 2;  // A col (floats)
  const int br0 = tid >> 3;         // B row (p=0), +32*p
  const int bs8 = (tid & 7) << 3;   // B col (bf16)

  v8f acc[4];
#pragma unroll
  for (int i = 0; i < 4; ++i)
#pragma unroll
    for (int j = 0; j < 8; ++j) acc[i][j] = 0.f;

  auto stage = [&](int buf, int kt) {
#pragma unroll
    for (int p = 0; p < 2; ++p) {
      const int r = p * 16 + ar0;
      const float* gp = A + (size_t)(rowBase + r) * GN + kt + ac;
#if HAVE_ASYNC_LDS
      async_cp16(gp, &As[buf][r][ac]);
#else
      *(float4*)&As[buf][r][ac] = *(const float4*)gp;
#endif
      __builtin_prefetch(gp + BK, 0, 0);  // one tile past the staged one
    }
#pragma unroll
    for (int p = 0; p < 8; ++p) {
      const int r = p * 32 + br0;
      const __bf16* gp = XsT + (size_t)r * GN + kt + bs8;
#if HAVE_ASYNC_LDS
      async_cp16(gp, &Bs[buf][r][bs8]);
#else
      *(uint4*)&Bs[buf][r][bs8] = *(const uint4*)gp;
#endif
    }
  };

  stage(0, 0);  // prologue: tile 0 in flight
  const int T = GN / BK;  // 128 K-tiles
  for (int t = 0; t < T; ++t) {
    const int cur = t & 1;
    if (t + 1 < T) {
      stage(cur ^ 1, (t + 1) * BK);  // issue next tile (10 async b128/thread)
      wait_async<10>();              // in-order: waits out current tile's copies
    } else {
      wait_async<0>();
    }
    __syncthreads();
#pragma unroll
    for (int kk = 0; kk < BK; kk += 32) {
      // A fragment (16x32 bf16): lane<16: M=lane, K=kk+0..15;
      //                          lane>=16: M=lane-16, K=kk+16..31
      const float* ap = &As[cur][wm * 16 + l15][kk + (hi << 4)];
      v16bf a;
#pragma unroll
      for (int i = 0; i < 8; ++i) {
        a[i]     = (__bf16)ap[i];
        a[i + 8] = (__bf16)ap[i + 8];
      }
#pragma unroll
      for (int nt = 0; nt < 4; ++nt) {
        // B fragment (32x16 bf16): lane<16: N=lane, K=kk+0..15;
        //                          lane>=16: N=lane-16, K=kk+16..31
        const __bf16* bp = &Bs[cur][wn * 64 + nt * 16 + l15][kk + (hi << 4)];
        v8bf b0 = *(const v8bf*)bp;
        v8bf b1 = *(const v8bf*)(bp + 8);
        v16bf b;
#pragma unroll
        for (int i = 0; i < 8; ++i) { b[i] = b0[i]; b[i + 8] = b1[i]; }
        acc[nt] = __builtin_amdgcn_wmma_f32_16x16x32_bf16(
            false, a, false, b, (short)0, acc[nt], false, false);
      }
    }
    __syncthreads();  // release buffer `cur` for restaging at t+2
  }
  // epilogue: C layout VGPR r -> M = r + 8*hi, N = l15; scale row, store bf16
  const int m0 = wm * 16 + hi * 8;
  const int c0 = wn * 64 + l15;
#pragma unroll
  for (int r = 0; r < 8; ++r) {
    const int grow = rowBase + m0 + r;
    const float s = dinv[grow];
#pragma unroll
    for (int nt = 0; nt < 4; ++nt)
      H[(size_t)grow * GD + c0 + nt * 16] = (__bf16)(acc[nt][r] * s);
  }
}

// ---------------- kernel 4: out = H @ W^T (f32 out) ----------------
__global__ __launch_bounds__(256) void k_proj(const __bf16* __restrict__ H,
                                              const __bf16* __restrict__ Wb,
                                              float* __restrict__ out) {
  __shared__ __bf16 As[BM][BK + 8];  // 32 x 72 bf16
  __shared__ __bf16 Bs[GD][BK + 8];  // 256 x 72 bf16
  const int tid  = threadIdx.x;
  const int lane = tid & 31;
  const int l15  = lane & 15;
  const int hi   = lane >> 4;
  const int wid  = tid >> 5;
  const int wm   = wid & 1;
  const int wn   = wid >> 1;
  const int rowBase = blockIdx.x * BM;

  v8f acc[4];
#pragma unroll
  for (int i = 0; i < 4; ++i)
#pragma unroll
    for (int j = 0; j < 8; ++j) acc[i][j] = 0.f;

  for (int kt = 0; kt < GD; kt += BK) {
    {
      const int r  = tid >> 3;
      const int s8 = (tid & 7) << 3;
      *(uint4*)&As[r][s8] = *(const uint4*)(H + (size_t)(rowBase + r) * GD + kt + s8);
    }
#pragma unroll
    for (int p = 0; p < 8; ++p) {
      const int idx = p * 256 + tid;
      const int r  = idx >> 3;
      const int s8 = (idx & 7) << 3;
      *(uint4*)&Bs[r][s8] = *(const uint4*)(Wb + (size_t)r * GD + kt + s8);
    }
    __syncthreads();
#pragma unroll
    for (int kk = 0; kk < BK; kk += 32) {
      const __bf16* ap = &As[wm * 16 + l15][kk + (hi << 4)];
      v8bf a0 = *(const v8bf*)ap;
      v8bf a1 = *(const v8bf*)(ap + 8);
      v16bf a;
#pragma unroll
      for (int i = 0; i < 8; ++i) { a[i] = a0[i]; a[i + 8] = a1[i]; }
#pragma unroll
      for (int nt = 0; nt < 4; ++nt) {
        const __bf16* bp = &Bs[wn * 64 + nt * 16 + l15][kk + (hi << 4)];
        v8bf b0 = *(const v8bf*)bp;
        v8bf b1 = *(const v8bf*)(bp + 8);
        v16bf b;
#pragma unroll
        for (int i = 0; i < 8; ++i) { b[i] = b0[i]; b[i + 8] = b1[i]; }
        acc[nt] = __builtin_amdgcn_wmma_f32_16x16x32_bf16(
            false, a, false, b, (short)0, acc[nt], false, false);
      }
    }
    __syncthreads();
  }
  const int m0 = wm * 16 + hi * 8;
  const int c0 = wn * 64 + l15;
#pragma unroll
  for (int r = 0; r < 8; ++r) {
    const int grow = rowBase + m0 + r;
#pragma unroll
    for (int nt = 0; nt < 4; ++nt)
      out[(size_t)grow * GD + c0 + nt * 16] = acc[nt][r];
  }
}

// ---------------- host launcher ----------------
extern "C" void kernel_launch(void* const* d_in, const int* in_sizes, int n_in,
                              void* d_out, int out_size, void* d_ws, size_t ws_size,
                              hipStream_t stream) {
  const float* X = (const float*)d_in[0];  // [8192, 256]
  const float* A = (const float*)d_in[1];  // [8192, 8192]
  const float* W = (const float*)d_in[2];  // [256, 256]
  float* out = (float*)d_out;              // [8192, 256] f32

  char* ws = (char*)d_ws;
  float*  dinv = (float*)ws;                                // 32 KB
  __bf16* Wb   = (__bf16*)(ws + 32768);                     // 128 KB
  __bf16* XsT  = (__bf16*)(ws + 32768 + 131072);            // 4 MB
  __bf16* H    = (__bf16*)(ws + 32768 + 131072 + 4194304);  // 4 MB
  (void)in_sizes; (void)n_in; (void)out_size; (void)ws_size;

  k_rowsum<<<GN, 256, 0, stream>>>(A, dinv);
  k_cvtW  <<<(GD * GD) / 256, 256, 0, stream>>>(W, Wb);
  k_scaleT<<<(GN * GD) / 256, 256, 0, stream>>>(X, dinv, XsT);
  k_aggr  <<<GN / BM, 256, 0, stream>>>(A, XsT, dinv, H);
  k_proj  <<<GN / BM, 256, 0, stream>>>(H, Wb, out);
}
